// SelfTensProds_16003048145202
// MI455X (gfx1250) — matrix operand order repaired and checked by reading, not compiled, not run on hardware
//
#include <hip/hip_runtime.h>
#include <stdint.h>

// ---------------------------------------------------------------------------
// SelfTensProds on MI455X (gfx1250): bf16 WMMA pipeline
//   P = 32*512 = 16384 rows, CHAN = 256, ranks 0/1/2 (26 l/r planes, 63 y planes)
// ---------------------------------------------------------------------------

#define P_ROWS   16384
#define NPLANES  26
#define EPS_GN   1e-5f

// k_cg_out geometry: 13 waves, one plane-group per wave, 16 N-tiles each
#define NW2      13
#define NT2      (NW2 * 32)   // 416 threads

typedef __attribute__((ext_vector_type(16))) __bf16 v16bf;
typedef __attribute__((ext_vector_type(8)))  float  v8f;

union FragB { v16bf v; uint32_t u[8]; };
union FragC { v8f   v; float    f[8]; };

__device__ __forceinline__ uint16_t f2bf(float f) {
  uint32_t u = __builtin_bit_cast(uint32_t, f);
  u += 0x7fffu + ((u >> 16) & 1u);          // round-to-nearest-even
  return (uint16_t)(u >> 16);
}
__device__ __forceinline__ float bf2f(uint16_t h) {
  uint32_t u = ((uint32_t)h) << 16;
  return __builtin_bit_cast(float, u);
}

// A-matrix 16x32 bf16 fragment from a row-major row of bf16 (viewed as u32 pairs).
// Lane L: row M = L&15. Lower half-lanes: K in {0..15}; upper: {16..31}.
__device__ __forceinline__ v16bf load_a_frag(const uint32_t* rowBase, int laneHi) {
  FragB f;
#pragma unroll
  for (int j = 0; j < 8; ++j) {
    int idx = j | (laneHi << 3);
    f.u[j] = rowBase[idx];
  }
  return f.v;
}

// B-matrix 32x16 bf16 fragment: lane L holds N = L&15, K = (L>=16?16:0)+0..15,
// i.e. 16 consecutive bf16 from W[n][kbase] -> two 16B loads.
__device__ __forceinline__ v16bf load_b_frag(const uint32_t* p) {
  FragB f;
  uint4 a = ((const uint4*)p)[0];
  uint4 b = ((const uint4*)p)[1];
  f.u[0] = a.x; f.u[1] = a.y; f.u[2] = a.z; f.u[3] = a.w;
  f.u[4] = b.x; f.u[5] = b.y; f.u[6] = b.z; f.u[7] = b.w;
  return f.v;
}

__device__ __forceinline__ v8f wmma_bf16(v16bf a, v16bf b, v8f c) {
  return __builtin_amdgcn_wmma_f32_16x16x32_bf16(false, a, false, b, (short)0, c,
                                                 false, false);
}

// ---------------------------------------------------------------------------
// Kernel 0: convert the 19 weight matrices (256x256 f32) to bf16 in workspace
// ---------------------------------------------------------------------------
struct WPtrs { const float* p[19]; };

__global__ void __launch_bounds__(256) k_convert_weights(WPtrs wp,
                                                         uint16_t* __restrict__ dst) {
  const int total = 19 * 256 * 256;
  for (int i = blockIdx.x * 256 + threadIdx.x; i < total; i += gridDim.x * 256) {
    int s = i >> 16;
    int off = i & 0xffff;
    dst[i] = f2bf(wp.p[s][off]);
  }
}

// ---------------------------------------------------------------------------
// Kernel 1: input TensLinears. x:[P][256][C] f32 -> l/r planes bf16
//   out layout: lr[(plane*P + row)*256 + outChan], plane in [0,26)
//   l0->0, r0->1, l1->2+i, r1->5+i, l2->8+ij, r2->17+ij
// ---------------------------------------------------------------------------
template <int C>
__global__ void __launch_bounds__(256) k_input_linear(
    const float* __restrict__ x, const uint16_t* __restrict__ wl,
    const uint16_t* __restrict__ wr, uint16_t* __restrict__ lr,
    int plane_l, int plane_r) {
  __shared__ __align__(16) uint16_t xs[C * 16 * 256];   // [c][row][i] bf16
  const int tid = threadIdx.x;
  const int rowbase = blockIdx.x * 16;

  // stage x tile -> LDS, transposing the component axis out
  const int total = 16 * 256 * C;
  const float* xg = x + (size_t)rowbase * 256 * C;
  for (int f = tid; f < total; f += 256) {
    int c = f % C;
    int t2 = f / C;
    int i = t2 & 255;
    int row = t2 >> 8;
    xs[(c * 16 + row) * 256 + i] = f2bf(xg[f]);
  }
  __syncthreads();

  // wave-uniform wave id -> SGPR, so job decode is scalar
  const int wid = __builtin_amdgcn_readfirstlane(tid >> 5);
  const int lane = tid & 31;
  const int laneHi = lane >> 4, lane15 = lane & 15;

  // loop-invariant per-lane pieces of the fragment addresses
  const int bLaneOff = lane15 * 256 + laneHi * 16;              // bf16 elements
  const uint32_t* aLaneBase = (const uint32_t*)xs + lane15 * 128;

  // 32 jobs = {L,R} x 16 N-tiles; 8 waves x 4 jobs
#pragma unroll 1
  for (int jj = 0; jj < 4; ++jj) {
    const int job = wid * 4 + jj;        // scalar
    const int lrSel = job >> 4;
    const int ntile = job & 15;
    const uint16_t* w = (lrSel ? wr : wl) + bLaneOff;
    FragC acc[C];
#pragma unroll
    for (int c = 0; c < C; ++c)
      acc[c].v = (v8f){0.f, 0.f, 0.f, 0.f, 0.f, 0.f, 0.f, 0.f};

#pragma unroll 1
    for (int kc = 0; kc < 8; ++kc) {            // K = 256 in chunks of 32
      v16bf bfr = load_b_frag((const uint32_t*)(w + ntile * 4096 + kc * 32));
#pragma unroll
      for (int c = 0; c < C; ++c) {
        v16bf afr = load_a_frag(aLaneBase + c * 2048 + kc * 16, laneHi);
        acc[c].v = wmma_bf16(afr, bfr, acc[c].v);
      }
    }

    const int planeBase = lrSel ? plane_r : plane_l;             // scalar
#pragma unroll
    for (int c = 0; c < C; ++c) {
#pragma unroll
      for (int j = 0; j < 8; ++j) {
        int m = laneHi * 8 + j;                 // C-frag: M = (L>>4)*8 + j
        int n = ntile * 16 + lane15;            //          N = L&15
        size_t off = ((size_t)(planeBase + c) * P_ROWS + rowbase + m) * 256 + n;
        lr[off] = f2bf(acc[c].f[j]);
      }
    }
  }
}

// ---------------------------------------------------------------------------
// Kernel 2: CG products + 13 output TensLinears + fused groupnorm.
// 416 threads = 13 waves. Wave w owns one "plane group":
//   w==0   : rank-0 outputs (3 pairs: y_000/110/220 x W_Y000/110/220)
//   w=1..3 : rank-1 comp c=w-1 (5 pairs)
//   w=4..12: rank-2 comp ij=w-4 (5 pairs)
// and all 16 N-tiles of that group (acc[16]); the A-fragment of each pair is
// loaded from LDS ONCE and reused for all 16 N-tiles, whose B addresses are
// compile-time immediate offsets off one scalar base.
// ---------------------------------------------------------------------------
__global__ void __launch_bounds__(NT2) k_cg_out(
    const uint16_t* __restrict__ lr, const uint16_t* __restrict__ wb,
    const float* __restrict__ gaw, const float* __restrict__ gab,
    const float* __restrict__ gvw, const float* __restrict__ gdw,
    float* __restrict__ out) {
  __shared__ __align__(16) uint16_t s_lr[NPLANES * 16 * 32];  // [plane][row][kk]
  __shared__ __align__(16) uint16_t s_y[63 * 16 * 32];        // [yplane][row][kk]
  __shared__ float sAsum[128], sAsq[128], sVsq[128], sDsq[128];  // [row][group]

  const int tid = threadIdx.x;
  const int rowbase = blockIdx.x * 16;
  for (int i = tid; i < 128; i += NT2) {
    sAsum[i] = 0.f; sAsq[i] = 0.f; sVsq[i] = 0.f; sDsq[i] = 0.f;
  }

  const int wid = __builtin_amdgcn_readfirstlane(tid >> 5);   // scalar wave id 0..12
  const int lane = tid & 31;
  const int laneHi = lane >> 4, lane15 = lane & 15;

  // loop-invariant per-lane address pieces
  const uint16_t* wLane = wb + lane15 * 256 + laneHi * 16;          // + scalar part
  const uint32_t* yLane = (const uint32_t*)s_y + lane15 * 16;       // + plane*256

  // --- scalar group decode, once ---
  int kind, comp;
  if (wid == 0)      { kind = 0; comp = 0; }
  else if (wid < 4)  { kind = 1; comp = wid - 1; }
  else               { kind = 2; comp = wid - 4; }
  int pl[5], wslot[5], np;
  if (kind == 0) {
    np = 3;
    pl[0] = 0; pl[1] = 1; pl[2] = 2; pl[3] = 0; pl[4] = 0;
    wslot[0] = 6; wslot[1] = 7; wslot[2] = 8; wslot[3] = 6; wslot[4] = 6;
  } else if (kind == 1) {
    np = 5;
    pl[0] = 3 + comp; pl[1] = 6 + comp; pl[2] = 9 + comp;
    pl[3] = 12 + comp; pl[4] = 15 + comp;
    wslot[0] = 9; wslot[1] = 10; wslot[2] = 11; wslot[3] = 12; wslot[4] = 13;
  } else {
    np = 5;
    pl[0] = 18 + comp; pl[1] = 27 + comp; pl[2] = 36 + comp;
    pl[3] = 45 + comp; pl[4] = 54 + comp;
    wslot[0] = 14; wslot[1] = 15; wslot[2] = 16; wslot[3] = 17; wslot[4] = 18;
  }

  FragC acc[16];
#pragma unroll
  for (int nt = 0; nt < 16; ++nt)
    acc[nt].v = (v8f){0.f, 0.f, 0.f, 0.f, 0.f, 0.f, 0.f, 0.f};

#pragma unroll 1
  for (int kc = 0; kc < 8; ++kc) {
    __syncthreads();
    // --- stage the 26 l/r planes for this K chunk (16 rows x 32 chans) ---
    for (int idx = tid; idx < NPLANES * 256; idx += NT2) {
      int p = idx >> 8;
      int row = (idx >> 4) & 15;
      int kkp = idx & 15;
      const uint32_t* src =
          (const uint32_t*)(lr + ((size_t)p * P_ROWS + rowbase + row) * 256 + kc * 32) +
          kkp;
      ((uint32_t*)s_lr)[idx] = *src;
    }
    __syncthreads();
    // --- compute all 63 CG y-planes elementwise ---
#pragma unroll 1
    for (int pos = tid; pos < 512; pos += NT2) {
      int row = pos >> 5, kk = pos & 31;
      float L0 = bf2f(s_lr[(0 * 16 + row) * 32 + kk]);
      float R0 = bf2f(s_lr[(1 * 16 + row) * 32 + kk]);
      float L1[3], R1[3], L2[9], R2[9];
#pragma unroll
      for (int i = 0; i < 3; ++i) {
        L1[i] = bf2f(s_lr[((2 + i) * 16 + row) * 32 + kk]);
        R1[i] = bf2f(s_lr[((5 + i) * 16 + row) * 32 + kk]);
      }
#pragma unroll
      for (int i = 0; i < 9; ++i) {
        L2[i] = bf2f(s_lr[((8 + i) * 16 + row) * 32 + kk]);
        R2[i] = bf2f(s_lr[((17 + i) * 16 + row) * 32 + kk]);
      }
      float Y[63];
      Y[0] = L0 * R0;
      Y[1] = L1[0] * R1[0] + L1[1] * R1[1] + L1[2] * R1[2];
      {
        float s = 0.f;
#pragma unroll
        for (int ij = 0; ij < 9; ++ij) s += L2[ij] * R2[ij];
        Y[2] = s;
      }
#pragma unroll
      for (int c = 0; c < 3; ++c) Y[3 + c] = L0 * R1[c];        // y_011
#pragma unroll
      for (int c = 0; c < 3; ++c) Y[6 + c] = L1[c] * R0;        // y_101
#pragma unroll
      for (int j = 0; j < 3; ++j)                               // y_121_j = sum_i l1_i r2_ij
        Y[9 + j] = L1[0] * R2[j] + L1[1] * R2[3 + j] + L1[2] * R2[6 + j];
#pragma unroll
      for (int i = 0; i < 3; ++i)                               // y_211_i = sum_j l2_ij r1_j
        Y[12 + i] = L2[3 * i] * R1[0] + L2[3 * i + 1] * R1[1] + L2[3 * i + 2] * R1[2];
      Y[15] = L1[1] * R1[2] - L1[2] * R1[1];                    // y_111 = l1 x r1
      Y[16] = L1[2] * R1[0] - L1[0] * R1[2];
      Y[17] = L1[0] * R1[1] - L1[1] * R1[0];
#pragma unroll
      for (int ij = 0; ij < 9; ++ij) Y[18 + ij] = L0 * R2[ij];  // y_022
#pragma unroll
      for (int ij = 0; ij < 9; ++ij) Y[27 + ij] = L2[ij] * R0;  // y_202
#pragma unroll
      for (int i = 0; i < 3; ++i)
#pragma unroll
        for (int j = 0; j < 3; ++j) Y[36 + 3 * i + j] = L1[i] * R1[j];  // y_112
#pragma unroll
      for (int i = 0; i < 3; ++i)                               // y_222_ij = sum_k l2_ik r2_kj
#pragma unroll
        for (int j = 0; j < 3; ++j)
          Y[45 + 3 * i + j] = L2[3 * i] * R2[j] + L2[3 * i + 1] * R2[3 + j] +
                              L2[3 * i + 2] * R2[6 + j];
#pragma unroll
      for (int i = 0; i < 3; ++i) {                             // y_212_ij = (l2_i x r1)_j
        float u0 = L2[3 * i], u1 = L2[3 * i + 1], u2 = L2[3 * i + 2];
        Y[54 + 3 * i + 0] = u1 * R1[2] - u2 * R1[1];
        Y[54 + 3 * i + 1] = u2 * R1[0] - u0 * R1[2];
        Y[54 + 3 * i + 2] = u0 * R1[1] - u1 * R1[0];
      }
      uint16_t* yb = s_y + row * 32 + kk;
#pragma unroll
      for (int p = 0; p < 63; ++p) yb[p * 512] = f2bf(Y[p]);
    }
    __syncthreads();
    // --- GEMM: per pair, load the A-frag once, sweep all 16 N-tiles ---
#pragma unroll 1
    for (int p = 0; p < 5; ++p) {
      if (p >= np) break;                               // scalar branch
      v16bf afr = load_a_frag(yLane + pl[p] * 256, laneHi);
      const uint16_t* wpB = wLane + (size_t)wslot[p] * 65536 + kc * 32;  // scalar base
#pragma unroll
      for (int nt = 0; nt < 16; ++nt) {                 // nt*4096 = immediate offset
        v16bf bfr = load_b_frag((const uint32_t*)(wpB + nt * 4096));
        acc[nt].v = wmma_bf16(afr, bfr, acc[nt].v);
      }
    }
  }

  // --- fused groupnorm: accumulate group stats in LDS (kind is scalar) ---
#pragma unroll
  for (int nt = 0; nt < 16; ++nt) {
    const int g = nt >> 1;  // 32-channel groups
#pragma unroll
    for (int j = 0; j < 8; ++j) {
      float v = acc[nt].f[j];
      int m = laneHi * 8 + j;
      if (kind == 0) {
        atomicAdd(&sAsum[m * 8 + g], v);
        atomicAdd(&sAsq[m * 8 + g], v * v);
      } else if (kind == 1) {
        atomicAdd(&sVsq[m * 8 + g], v * v);
      } else {
        atomicAdd(&sDsq[m * 8 + g], v * v);
      }
    }
  }
  __syncthreads();
  if (tid < 128) {
    float mean = sAsum[tid] * (1.f / 32.f);
    float var = sAsq[tid] * (1.f / 32.f) - mean * mean;
    sAsum[tid] = mean;
    sAsq[tid] = rsqrtf(var + EPS_GN);
    sVsq[tid] = rsqrtf(sVsq[tid] * (1.f / 96.f) + EPS_GN);   // 32 ch * 3 comps
    sDsq[tid] = rsqrtf(sDsq[tid] * (1.f / 288.f) + EPS_GN);  // 32 ch * 9 comps
  }
  __syncthreads();

  const size_t OFFV = (size_t)P_ROWS * 256;
  const size_t OFFD = (size_t)P_ROWS * 256 * 4;
#pragma unroll
  for (int nt = 0; nt < 16; ++nt) {
    const int g = nt >> 1;
    const int n = nt * 16 + lane15;
#pragma unroll
    for (int j = 0; j < 8; ++j) {
      float v = acc[nt].f[j];
      int m = laneHi * 8 + j;
      size_t rg = (size_t)(rowbase + m);
      if (kind == 0) {
        float o = (v - sAsum[m * 8 + g]) * sAsq[m * 8 + g] * gaw[n] + gab[n];
        out[rg * 256 + n] = o;
      } else if (kind == 1) {
        out[OFFV + (rg * 256 + n) * 3 + comp] = v * sVsq[m * 8 + g] * gvw[n];
      } else {
        out[OFFD + (rg * 256 + n) * 9 + comp] = v * sDsq[m * 8 + g] * gdw[n];
      }
    }
  }
}

// ---------------------------------------------------------------------------
// Launch
//   d_in: 0 x_a, 1 x_v, 2 x_d, 3..8 W_L0,W_R0,W_L1,W_R1,W_L2,W_R2,
//         9..21 W_Y000,W_Y110,W_Y220,W_Y011,W_Y101,W_Y121,W_Y211,W_Y111,
//               W_Y022,W_Y202,W_Y112,W_Y222,W_Y212,
//         22 gn_a_w, 23 gn_a_b, 24 gn_v_w, 25 gn_d_w
//   ws: [0, 26*P*256) bf16 l/r planes; then 19*65536 bf16 converted weights
// ---------------------------------------------------------------------------
extern "C" void kernel_launch(void* const* d_in, const int* in_sizes, int n_in,
                              void* d_out, int out_size, void* d_ws, size_t ws_size,
                              hipStream_t stream) {
  (void)in_sizes; (void)n_in; (void)out_size; (void)ws_size;
  const float* x_a = (const float*)d_in[0];
  const float* x_v = (const float*)d_in[1];
  const float* x_d = (const float*)d_in[2];

  uint16_t* lr = (uint16_t*)d_ws;
  uint16_t* wb = lr + (size_t)NPLANES * P_ROWS * 256;

  WPtrs wp;
  for (int s = 0; s < 19; ++s) wp.p[s] = (const float*)d_in[3 + s];
  k_convert_weights<<<1024, 256, 0, stream>>>(wp, wb);

  const int nblk = P_ROWS / 16;  // 1024
  k_input_linear<1><<<nblk, 256, 0, stream>>>(x_a, wb + 0 * 65536, wb + 1 * 65536, lr, 0, 1);
  k_input_linear<3><<<nblk, 256, 0, stream>>>(x_v, wb + 2 * 65536, wb + 3 * 65536, lr, 2, 5);
  k_input_linear<9><<<nblk, 256, 0, stream>>>(x_d, wb + 4 * 65536, wb + 5 * 65536, lr, 8, 17);

  k_cg_out<<<nblk, NT2, 0, stream>>>(lr, wb, (const float*)d_in[22],
                                     (const float*)d_in[23], (const float*)d_in[24],
                                     (const float*)d_in[25], (float*)d_out);
}
